// Attention_44710609551920
// MI455X (gfx1250) — matrix operand order repaired and checked by reading, not compiled, not run on hardware
//
#include <hip/hip_runtime.h>

// MHA forward for B=2, S=4096, H=4, Dh=32, D=128 on gfx1250 (wave32 WMMA).
// All matmuls use v_wmma_f32_16x16x32_f16; softmax + blend done in f32.
// k_attn processes TWO query tiles per wave so every K/V^T fragment load
// feeds 2x the WMMA work (L2-bandwidth roofline driven choice).

typedef __attribute__((ext_vector_type(16))) _Float16 v16h;
typedef __attribute__((ext_vector_type(8)))  _Float16 v8h;
typedef __attribute__((ext_vector_type(8)))  float    v8f;
typedef __attribute__((ext_vector_type(4)))  float    f4;

#define D_MODEL   128
#define NUM_HEADS 4
#define D_HEAD    32
#define SEQ       4096
#define BATCH     2
#define QK_SCALE  0.17677669529663687f  // 1/sqrt(32)

static __device__ __forceinline__ v8f wmma_f16(v16h a, v16h b, v8f c) {
  // (neg_a, A, neg_b, B, c_mod, C, reuse_a, reuse_b)
  return __builtin_amdgcn_wmma_f32_16x16x32_f16(false, a, false, b, (short)0, c,
                                                false, false);
}

// ---------------------------------------------------------------------------
// Kernel 1: fused QKV projection. C[16x32] = X[16x128] * W[h]^T per wave-task.
// Q is pre-scaled by 1/sqrt(Dh). V is stored transposed: Vt[b][h][dh][s].
// A frag (16x32 f16): lane(row=l&15, hh=l>>4); a[i]=x[row][kk+16hh+i] (one contiguous run).
// B frag (32x16 f16): lane(col=n); b[i]=W[n][kk+16hh+i] (one contiguous run).
// ---------------------------------------------------------------------------
__global__ __launch_bounds__(256) void k_qkv(const float* __restrict__ x,
                                             const float* __restrict__ Wk,
                                             const float* __restrict__ Wq,
                                             const float* __restrict__ Wv,
                                             _Float16* __restrict__ Qb,
                                             _Float16* __restrict__ Kb,
                                             _Float16* __restrict__ Vt) {
  const int lane = threadIdx.x & 31;
  const int wave = threadIdx.x >> 5;
  const int row  = lane & 15;
  const int hh   = lane >> 4;

  const int task = blockIdx.x * 8 + wave;   // 0..6143
  const int mt   = task & 511;              // 512 row tiles of 16 tokens
  const int h    = (task >> 9) & 3;
  const int mat  = task >> 11;              // 0=K, 1=Q, 2=V
  const float* W =
      (mat == 0 ? Wk : (mat == 1 ? Wq : Wv)) + (size_t)h * D_HEAD * D_MODEL;
  const int m0 = mt * 16;

  const float* xrow  = x + (size_t)(m0 + row) * D_MODEL;
  const float* wrow0 = W + (size_t)row * D_MODEL;         // dh = row
  const float* wrow1 = W + (size_t)(16 + row) * D_MODEL;  // dh = 16+row

  v8f c0 = {}; v8f c1 = {};
#pragma unroll
  for (int kk = 0; kk < D_MODEL; kk += 32) {
    v16h a;
    {
      f4 p0 = *(const f4*)(xrow + kk + 16 * hh);
      f4 p1 = *(const f4*)(xrow + kk + 16 * hh + 4);
      f4 p2 = *(const f4*)(xrow + kk + 16 * hh + 8);
      f4 p3 = *(const f4*)(xrow + kk + 16 * hh + 12);
#pragma unroll
      for (int i = 0; i < 4; ++i) {
        a[i]      = (_Float16)p0[i];
        a[4 + i]  = (_Float16)p1[i];
        a[8 + i]  = (_Float16)p2[i];
        a[12 + i] = (_Float16)p3[i];
      }
    }
    v16h b0, b1;
#pragma unroll
    for (int i = 0; i < 16; i += 4) {
      f4 w0 = *(const f4*)(wrow0 + kk + 16 * hh + i);
      f4 w1 = *(const f4*)(wrow1 + kk + 16 * hh + i);
#pragma unroll
      for (int j = 0; j < 4; ++j) {
        b0[i + j] = (_Float16)w0[j];
        b1[i + j] = (_Float16)w1[j];
      }
    }
    c0 = wmma_f16(a, b0, c0);
    c1 = wmma_f16(a, b1, c1);
  }

  // D frag: lane col n = row; vgpr r -> token row m0 + r + 8*hh.
  if (mat == 2) {
    // Vt[(bh*32 + dh)*SEQ + s]; tokens r=0..7 are consecutive -> b128 store.
    const int tok0 = m0 + 8 * hh;
    const int bb   = tok0 >> 12;
    const int s0   = tok0 & 4095;
    const int bh   = bb * NUM_HEADS + h;
    v8h hv0, hv1;
#pragma unroll
    for (int r = 0; r < 8; ++r) {
      hv0[r] = (_Float16)c0[r];
      hv1[r] = (_Float16)c1[r];
    }
    *(v8h*)(Vt + ((size_t)(bh * D_HEAD + row) * SEQ + s0))      = hv0;
    *(v8h*)(Vt + ((size_t)(bh * D_HEAD + 16 + row) * SEQ + s0)) = hv1;
  } else {
    _Float16* dst = (mat == 1) ? Qb : Kb;
    const float sc = (mat == 1) ? QK_SCALE : 1.0f;
#pragma unroll
    for (int r = 0; r < 8; ++r) {
      const int tok = m0 + r + 8 * hh;
      const int bb  = tok >> 12;
      const int s   = tok & 4095;
      const int bh  = bb * NUM_HEADS + h;
      const size_t base = ((size_t)(bh * SEQ + s)) * D_HEAD;
      dst[base + row]      = (_Float16)(c0[r] * sc);
      dst[base + 16 + row] = (_Float16)(c1[r] * sc);
    }
  }
}

// ---------------------------------------------------------------------------
// Kernel 2: vsum[bh][dh] = sum_p V[p][dh]  (for the 0.5*uniform blend term)
// ---------------------------------------------------------------------------
__global__ __launch_bounds__(256) void k_vsum(const _Float16* __restrict__ Vt,
                                              float* __restrict__ vsum) {
  __shared__ float red[256];
  const int bh = blockIdx.x;
  const int dh = threadIdx.x & 31;
  const int ch = threadIdx.x >> 5;
  const _Float16* src = Vt + ((size_t)(bh * D_HEAD + dh)) * SEQ + ch * 512;
  float s = 0.f;
  for (int i = 0; i < 512; i += 8) {
    v8h v = *(const v8h*)(src + i);
#pragma unroll
    for (int j = 0; j < 8; ++j) s += (float)v[j];
  }
  red[threadIdx.x] = s;
  __syncthreads();
  if (ch == 0) {
    float t = 0.f;
#pragma unroll
    for (int c = 0; c < 8; ++c) t += red[dh + 32 * c];
    vsum[bh * D_HEAD + dh] = t;
  }
}

// ---------------------------------------------------------------------------
// Online-softmax step on one S^T tile pair (rows = keys p, cols = queries q).
// Per-query reductions are in-lane over 8 D-VGPRs + one shfl_xor(16).
// Emits the P^T B-fragment (k = key offset 0..31) via half-swap shuffles only.
// ---------------------------------------------------------------------------
static __device__ __forceinline__ void softmax_step(v8f s0, v8f s1, int hh,
                                                    float& mrun, float& lrun,
                                                    v8f& o0, v8f& o1,
                                                    v16h& pb) {
  float cmax = s0[0];
#pragma unroll
  for (int r = 0; r < 8; ++r) {
    cmax = fmaxf(cmax, s0[r]);
    cmax = fmaxf(cmax, s1[r]);
  }
  cmax = fmaxf(cmax, __shfl_xor(cmax, 16, 32));
  const float mnew  = fmaxf(mrun, cmax);
  const float alpha = __expf(mrun - mnew);
  mrun = mnew;

  float e0[8], e1[8];
  float rs = 0.f;
#pragma unroll
  for (int r = 0; r < 8; ++r) {
    e0[r] = __expf(s0[r] - mnew);
    e1[r] = __expf(s1[r] - mnew);
    rs += e0[r] + e1[r];
  }
  rs += __shfl_xor(rs, 16, 32);
  lrun = lrun * alpha + rs;
#pragma unroll
  for (int r = 0; r < 8; ++r) { o0[r] *= alpha; o1[r] *= alpha; }

  // lanes<16 need keys 0..15 (tile0), lanes>=16 need keys 16..31 (tile1).
#pragma unroll
  for (int r = 0; r < 8; ++r) {
    const float y0 = __shfl_xor(e0[r], 16, 32);
    const float y1 = __shfl_xor(e1[r], 16, 32);
    pb[r]     = (_Float16)(hh ? y1 : e0[r]);
    pb[8 + r] = (_Float16)(hh ? e1[r] : y0);
  }
}

// ---------------------------------------------------------------------------
// Kernel 3: flash attention, TWO 16-query tiles per wave (N=32):
//   S^T tiles = K_tile(16p x 32dh) x Q^T(32dh x 16q)   [4 WMMA / 32 keys]
//   Z^T tiles = Vt_tile(16dh x 32p) x P^T(32p x 16q)   [4 WMMA / 32 keys]
// K/V^T fragments are loaded once per chunk and reused by both query groups
// (2x arithmetic intensity vs one tile/wave -> halves L2 fragment traffic).
//   Final: z = 0.5 * O/l + 0.5 * vsum  (post-softmax affine blend, exact).
// ---------------------------------------------------------------------------
__global__ __launch_bounds__(256) void k_attn(const _Float16* __restrict__ Qb,
                                              const _Float16* __restrict__ Kb,
                                              const _Float16* __restrict__ Vt,
                                              const float* __restrict__ vsum,
                                              _Float16* __restrict__ Z) {
  const int lane = threadIdx.x & 31;
  const int wave = threadIdx.x >> 5;
  const int row  = lane & 15;
  const int hh   = lane >> 4;

  const int task = blockIdx.x * 8 + wave;  // 0..1023
  const int qt   = task & 127;             // 128 query groups of 32
  const int bh   = task >> 7;
  const int q0   = qt * 32;

  const _Float16* Qp = Qb + (size_t)bh * SEQ * D_HEAD;
  const _Float16* Kp = Kb + (size_t)bh * SEQ * D_HEAD;
  const _Float16* Vp = Vt + (size_t)bh * D_HEAD * SEQ;

  // Q^T B-fragments (k=dh) for both query tiles (pre-scaled by 1/sqrt(Dh)).
  const v16h qfA = *(const v16h*)(Qp + (size_t)(q0 + row) * D_HEAD + 16 * hh);
  const v16h qfB =
      *(const v16h*)(Qp + (size_t)(q0 + 16 + row) * D_HEAD + 16 * hh);

  v8f o0a = {}, o1a = {}, o0b = {}, o1b = {};
  float mA = -3.0e38f, lA = 0.f;
  float mB = -3.0e38f, lB = 0.f;
  const v8f zero = {};

  for (int p0 = 0; p0 < SEQ; p0 += 32) {
    const _Float16* k0r = Kp + (size_t)(p0 + row) * D_HEAD;
    const _Float16* k1r = Kp + (size_t)(p0 + 16 + row) * D_HEAD;
    const _Float16* v0r = Vp + (size_t)row * SEQ + p0;
    const _Float16* v1r = Vp + (size_t)(16 + row) * SEQ + p0;
    if (p0 + 32 < SEQ) {  // gfx1250 global_prefetch_b8 for next chunk
      __builtin_prefetch(k0r + 32 * D_HEAD, 0, 3);
      __builtin_prefetch(v0r + 32, 0, 3);
    }

    // K A-fragments (rows = keys, k = dh)
    v16h ka0, ka1;
    {
      v8h x0 = *(const v8h*)(k0r + 16 * hh);
      v8h x1 = *(const v8h*)(k0r + 16 * hh + 8);
      v8h y0 = *(const v8h*)(k1r + 16 * hh);
      v8h y1 = *(const v8h*)(k1r + 16 * hh + 8);
#pragma unroll
      for (int i = 0; i < 8; ++i) {
        ka0[i] = x0[i]; ka0[8 + i] = x1[i];
        ka1[i] = y0[i]; ka1[8 + i] = y1[i];
      }
    }
    // V^T A-fragments (rows = dh, k = key offset), p-contiguous loads.
    v16h va0, va1;
    {
      v8h x0 = *(const v8h*)(v0r + 16 * hh);
      v8h x1 = *(const v8h*)(v0r + 16 * hh + 8);
      v8h y0 = *(const v8h*)(v1r + 16 * hh);
      v8h y1 = *(const v8h*)(v1r + 16 * hh + 8);
#pragma unroll
      for (int i = 0; i < 8; ++i) {
        va0[i] = x0[i]; va0[8 + i] = x1[i];
        va1[i] = y0[i]; va1[8 + i] = y1[i];
      }
    }

    v8f s0a = wmma_f16(ka0, qfA, zero);  // S^T rows p0..p0+15, queries A
    v8f s1a = wmma_f16(ka1, qfA, zero);  // S^T rows p0+16..p0+31, queries A
    v8f s0b = wmma_f16(ka0, qfB, zero);  // queries B
    v8f s1b = wmma_f16(ka1, qfB, zero);

    v16h pbA, pbB;
    softmax_step(s0a, s1a, hh, mA, lA, o0a, o1a, pbA);
    softmax_step(s0b, s1b, hh, mB, lB, o0b, o1b, pbB);

    o0a = wmma_f16(va0, pbA, o0a);  // Z^T rows dh 0..15, queries A
    o1a = wmma_f16(va1, pbA, o1a);  // Z^T rows dh 16..31, queries A
    o0b = wmma_f16(va0, pbB, o0b);  // queries B
    o1b = wmma_f16(va1, pbB, o1b);
  }

  // Epilogue: z = 0.5 * O / l + 0.5 * vsum ; write Z as f16 [token][h*32+dh].
  const float* vs = vsum + bh * D_HEAD;
  f4 u0a = *(const f4*)(vs + 8 * hh);
  f4 u0b = *(const f4*)(vs + 8 * hh + 4);
  f4 w0a = *(const f4*)(vs + 16 + 8 * hh);
  f4 w0b = *(const f4*)(vs + 16 + 8 * hh + 4);
  float uu[8], ww[8];
#pragma unroll
  for (int j = 0; j < 4; ++j) {
    uu[j] = u0a[j]; uu[4 + j] = u0b[j];
    ww[j] = w0a[j]; ww[4 + j] = w0b[j];
  }
  const int bb = bh >> 2;
  const int h  = bh & 3;

  {
    const float inv = 0.5f / lA;
    v8h z0, z1;
#pragma unroll
    for (int r = 0; r < 8; ++r) {
      z0[r] = (_Float16)(o0a[r] * inv + 0.5f * uu[r]);
      z1[r] = (_Float16)(o1a[r] * inv + 0.5f * ww[r]);
    }
    const int tok = (bb << 12) + q0 + row;  // D col = query (group A)
    const size_t zbase = (size_t)tok * (NUM_HEADS * D_HEAD) + h * D_HEAD;
    *(v8h*)(Z + zbase + 8 * hh)      = z0;
    *(v8h*)(Z + zbase + 16 + 8 * hh) = z1;
  }
  {
    const float inv = 0.5f / lB;
    v8h z0, z1;
#pragma unroll
    for (int r = 0; r < 8; ++r) {
      z0[r] = (_Float16)(o0b[r] * inv + 0.5f * uu[r]);
      z1[r] = (_Float16)(o1b[r] * inv + 0.5f * ww[r]);
    }
    const int tok = (bb << 12) + q0 + 16 + row;  // D col = query (group B)
    const size_t zbase = (size_t)tok * (NUM_HEADS * D_HEAD) + h * D_HEAD;
    *(v8h*)(Z + zbase + 8 * hh)      = z0;
    *(v8h*)(Z + zbase + 16 + 8 * hh) = z1;
  }
}

// ---------------------------------------------------------------------------
// Kernel 4: output projection Out[8192x128] = Z[8192x128] * W_O^T (f32 out)
// ---------------------------------------------------------------------------
__global__ __launch_bounds__(256) void k_oproj(const _Float16* __restrict__ Z,
                                               const float* __restrict__ Wo,
                                               float* __restrict__ out) {
  const int lane = threadIdx.x & 31;
  const int wave = threadIdx.x >> 5;
  const int row  = lane & 15;
  const int hh   = lane >> 4;

  const int task = blockIdx.x * 8 + wave;  // 0..2047
  const int mt   = task & 511;
  const int ng   = task >> 9;              // 4 groups of 32 output cols
  const int m0 = mt * 16, n0 = ng * 32;

  const _Float16* zrow = Z + (size_t)(m0 + row) * D_MODEL;
  const float* w0r = Wo + (size_t)(n0 + row) * D_MODEL;
  const float* w1r = Wo + (size_t)(n0 + 16 + row) * D_MODEL;

  v8f c0 = {}, c1 = {};
#pragma unroll
  for (int kk = 0; kk < D_MODEL; kk += 32) {
    v16h a;
    {
      v8h x0 = *(const v8h*)(zrow + kk + 16 * hh);
      v8h x1 = *(const v8h*)(zrow + kk + 16 * hh + 8);
#pragma unroll
      for (int i = 0; i < 8; ++i) { a[i] = x0[i]; a[8 + i] = x1[i]; }
    }
    v16h b0, b1;
#pragma unroll
    for (int i = 0; i < 16; i += 4) {
      f4 wv0 = *(const f4*)(w0r + kk + 16 * hh + i);
      f4 wv1 = *(const f4*)(w1r + kk + 16 * hh + i);
#pragma unroll
      for (int j = 0; j < 4; ++j) {
        b0[i + j] = (_Float16)wv0[j];
        b1[i + j] = (_Float16)wv1[j];
      }
    }
    c0 = wmma_f16(a, b0, c0);
    c1 = wmma_f16(a, b1, c1);
  }
#pragma unroll
  for (int r = 0; r < 8; ++r) {
    const int tok = m0 + r + 8 * hh;
    out[(size_t)tok * D_MODEL + n0 + row]      = c0[r];
    out[(size_t)tok * D_MODEL + n0 + 16 + row] = c1[r];
  }
}

extern "C" void kernel_launch(void* const* d_in, const int* in_sizes, int n_in,
                              void* d_out, int out_size, void* d_ws,
                              size_t ws_size, hipStream_t stream) {
  (void)in_sizes; (void)n_in; (void)out_size; (void)ws_size;
  const float* x  = (const float*)d_in[0];
  const float* Wk = (const float*)d_in[1];
  const float* Wq = (const float*)d_in[2];
  const float* Wv = (const float*)d_in[3];
  const float* Wo = (const float*)d_in[4];

  char* ws = (char*)d_ws;  // ~8 MB + 1 KB of f16 intermediates (L2-resident)
  _Float16* Qb = (_Float16*)(ws + (0u << 20));
  _Float16* Kb = (_Float16*)(ws + (2u << 20));
  _Float16* Vt = (_Float16*)(ws + (4u << 20));
  _Float16* Zb = (_Float16*)(ws + (6u << 20));
  float*  vsum = (float*)(ws + (8u << 20));

  k_qkv <<<768, 256, 0, stream>>>(x, Wk, Wq, Wv, Qb, Kb, Vt);
  k_vsum<<<  8, 256, 0, stream>>>(Vt, vsum);
  k_attn<<<128, 256, 0, stream>>>(Qb, Kb, Vt, vsum, Zb);
  k_oproj<<<256, 256, 0, stream>>>(Zb, Wo, (float*)d_out);
}
